// StateSpaceModel_5454608466171
// MI455X (gfx1250) — matrix operand order repaired and checked by reading, not compiled, not run on hardware
//
#include <hip/hip_runtime.h>
#include <hip/hip_bf16.h>

typedef __attribute__((ext_vector_type(16))) _Float16 v16h;
typedef __attribute__((ext_vector_type(8)))  _Float16 v8h;
typedef __attribute__((ext_vector_type(8)))  float    v8f;

#define ZD   64
#define XD   32
#define THD  128
#define OHD  64
#define TLEN 256
#define NBATCH 4096

// ---- packed f16 weight offsets (in halves), all [N_out][K_in] row-major ----
#define OFF_GRU   0        // [128][128]  rows: r(0..63),u(64..127); cols: [z|h]
#define OFF_IN    16384    // [64][64]    W_ih n-gate rows
#define OFF_HN    20480    // [64][64]    W_hh n-gate rows
#define OFF_GZH   24576    // [128][64]
#define OFF_GHZ   32768    // [64][128]
#define OFF_PZH   40960    // [128][64]
#define OFF_PHZ   49152    // [64][128]
#define OFF_ZLOC  57344    // [64][64]
#define OFF_OLH   61440    // [64][64]
#define OFF_OLX   65536    // [32][64]
#define OFF_OSH   67584    // [64][64]
#define OFF_OSX   71680    // [32][64]
#define NWH       73728

// ---- f32 bias offsets (floats), region after the halves ----
#define BOFF_RU   0     // 128 (b_ih+b_hh for r,u)
#define BOFF_IN   128   // 64
#define BOFF_HN   192   // 64
#define BOFF_GZH  256   // 128
#define BOFF_GHZ  384   // 64
#define BOFF_PZH  448   // 128
#define BOFF_PHZ  576   // 64
#define BOFF_ZLOC 640   // 64
#define BOFF_OLH  704   // 64
#define BOFF_OLX  768   // 32
#define BOFF_OSH  800   // 64
#define BOFF_OSX  864   // 32
#define NBF       896

__device__ __forceinline__ float sigf(float x)  { return 1.0f / (1.0f + __expf(-x)); }
__device__ __forceinline__ float tanhf_(float x){ float e = __expf(2.0f * x); return (e - 1.0f) / (e + 1.0f); }

// WMMA B operand: weight [N][K] f16 row-major in LDS.
// lane L supplies column n = ntile*16 + L%16, K rows kb*32 + (L/16)*8 .. +7 and kb*32 + 16 + (L/16)*8 .. +7
__device__ __forceinline__ v16h ldB(const _Float16* W, int Kdim, int ntile, int kb, int lane) {
    const _Float16* p = W + (size_t)(ntile * 16 + (lane & 15)) * Kdim + kb * 32 + ((lane >> 4) << 3);
    union { v16h v; v8h h[2]; } u;
    u.h[0] = *(const v8h*)p;
    u.h[1] = *(const v8h*)(p + 16);
    return u.v;
}

// WMMA A operand from row-major [16][Kdim] f16 activation in LDS.
// lane L: row m = L%16; halves 0..7 -> k = khalf..+7, halves 8..15 -> k = khalf+16..+23,
// khalf = kb*32 + (L>=16 ? 8 : 0)
__device__ __forceinline__ v16h ldA(const _Float16* act, int Kdim, int kb, int lane) {
    const _Float16* p = act + (size_t)(lane & 15) * Kdim + kb * 32 + ((lane >> 4) ? 8 : 0);
    union { v16h v; v8h h[2]; } u;
    u.h[0] = *(const v8h*)p;
    u.h[1] = *(const v8h*)(p + 16);
    return u.v;
}

// Store a 16x16 f32 C/D tile into row-major f16 activation buffer.
// lane L holds col n = L%16, rows m = i + 8*(L/16)
__device__ __forceinline__ void stAct(_Float16* act, int Kdim, int ntile, v8f c, int lane) {
    _Float16* p = act + (size_t)((lane >> 4) * 8) * Kdim + ntile * 16 + (lane & 15);
#pragma unroll
    for (int i = 0; i < 8; ++i) p[(size_t)i * Kdim] = (_Float16)c[i];
}

// bias broadcast into a C/D accumulator tile
__device__ __forceinline__ v8f bbc(const float* wb, int off, int ntile, int lane) {
    float b = wb[off + ntile * 16 + (lane & 15)];
    v8f c;
#pragma unroll
    for (int i = 0; i < 8; ++i) c[i] = b;
    return c;
}

#define WMMA(c, a, b) \
    c = __builtin_amdgcn_wmma_f32_16x16x32_f16(false, (a), false, (b), (short)0, (c), false, false)

// ---------------- weight packing: f32 params -> f16 [N][K] + fused biases ----------------
__global__ void pack_weights(
    const float* Wih, const float* Whh, const float* bih, const float* bhh,
    const float* Wgzh, const float* bgzh, const float* Wghz, const float* bghz,
    const float* Wpzh, const float* bpzh, const float* Wphz, const float* bphz,
    const float* Wzloc, const float* bzloc,
    const float* Wolh, const float* bolh, const float* Wolx, const float* bolx,
    const float* Wosh, const float* bosh, const float* Wosx, const float* bosx,
    _Float16* wh, float* wb)
{
    int i = blockIdx.x * blockDim.x + threadIdx.x;
    if (i < NWH) {
        float v;
        if (i < OFF_IN)        { int r = i >> 7, k = i & 127;
                                 v = (k < 64) ? Wih[r * 64 + k] : Whh[r * 64 + (k - 64)]; }
        else if (i < OFF_HN)   { int j = i - OFF_IN;  int r = j >> 6, k = j & 63; v = Wih[(128 + r) * 64 + k]; }
        else if (i < OFF_GZH)  { int j = i - OFF_HN;  int r = j >> 6, k = j & 63; v = Whh[(128 + r) * 64 + k]; }
        else if (i < OFF_GHZ)  { v = Wgzh[i - OFF_GZH]; }
        else if (i < OFF_PZH)  { v = Wghz[i - OFF_GHZ]; }
        else if (i < OFF_PHZ)  { v = Wpzh[i - OFF_PZH]; }
        else if (i < OFF_ZLOC) { v = Wphz[i - OFF_PHZ]; }
        else if (i < OFF_OLH)  { v = Wzloc[i - OFF_ZLOC]; }
        else if (i < OFF_OLX)  { v = Wolh[i - OFF_OLH]; }
        else if (i < OFF_OSH)  { v = Wolx[i - OFF_OLX]; }
        else if (i < OFF_OSX)  { v = Wosh[i - OFF_OSH]; }
        else                   { v = Wosx[i - OFF_OSX]; }
        wh[i] = (_Float16)v;
    }
    int j = i - NWH;
    if (j >= 0 && j < NBF) {
        float v;
        if (j < BOFF_IN)        v = bih[j] + bhh[j];
        else if (j < BOFF_HN)   v = bih[128 + (j - BOFF_IN)];
        else if (j < BOFF_GZH)  v = bhh[128 + (j - BOFF_HN)];
        else if (j < BOFF_GHZ)  v = bgzh[j - BOFF_GZH];
        else if (j < BOFF_PZH)  v = bghz[j - BOFF_GHZ];
        else if (j < BOFF_PHZ)  v = bpzh[j - BOFF_PZH];
        else if (j < BOFF_ZLOC) v = bphz[j - BOFF_PHZ];
        else if (j < BOFF_OLH)  v = bzloc[j - BOFF_ZLOC];
        else if (j < BOFF_OLX)  v = bolh[j - BOFF_OLH];
        else if (j < BOFF_OSH)  v = bolx[j - BOFF_OLX];
        else if (j < BOFF_OSX)  v = bosh[j - BOFF_OSH];
        else                    v = bosx[j - BOFF_OSX];
        wb[j] = v;
    }
}

// ---------------- main rollout: 2 waves/block, each wave owns 16 batch rows ----------------
__global__ void __launch_bounds__(64)
dmm_rollout(const float* __restrict__ z0, const float* __restrict__ h0,
            const _Float16* __restrict__ wh, const float* __restrict__ wb,
            float* __restrict__ out)
{
    __shared__ __align__(32) _Float16 sW[NWH];          // 144 KB weights (f16)
    __shared__ __align__(32) _Float16 sAct[2][16 * THD]; // per-wave bounce, 16x128 max

    const int tid  = threadIdx.x;
    const int lane = tid & 31;
    const int wv   = tid >> 5;

    // cooperative weight stage: global -> LDS (16B chunks)
    {
        const uint4* src = (const uint4*)wh;
        uint4*       dst = (uint4*)sW;
        for (int i = tid; i < NWH / 8; i += 64) dst[i] = src[i];
    }
    __syncthreads();

    _Float16* act = sAct[wv];
    const int rowBase = blockIdx.x * 32 + wv * 16;

    // ---- recurrent state ----
    v16h zA[2], hA[2];  // A-layout f16 (K=64)
    v8f  hD[4];         // D-layout f32 h_prev (16x64)
    {
        const int khalf = (lane >> 4) ? 8 : 0;
#pragma unroll
        for (int kb = 0; kb < 2; ++kb) {
#pragma unroll
            for (int j = 0; j < 16; ++j) {
                int k = kb * 32 + khalf + (j < 8 ? j : j + 8);
                zA[kb][j] = (_Float16)z0[k];
                hA[kb][j] = (_Float16)h0[k];
            }
        }
#pragma unroll
        for (int n4 = 0; n4 < 4; ++n4) {
            float hv = h0[n4 * 16 + (lane & 15)];
#pragma unroll
            for (int i = 0; i < 8; ++i) hD[n4][i] = hv;
        }
    }

#pragma unroll 1
    for (int t = 0; t < TLEN; ++t) {
        // ================= GRUCell =================
        // fused [z|h] x [Wih_ru;Whh_ru] gives i_r+h_r, i_u+h_u ; i_n, h_n separate
#pragma unroll
        for (int j = 0; j < 4; ++j) {
            v8f r = bbc(wb, BOFF_RU, j, lane);
            v8f u = bbc(wb, BOFF_RU, j + 4, lane);
#pragma unroll
            for (int kb = 0; kb < 4; ++kb) {
                v16h aop = (kb < 2) ? zA[kb] : hA[kb - 2];
                WMMA(r, aop, ldB(sW + OFF_GRU, 128, j,     kb, lane));
                WMMA(u, aop, ldB(sW + OFF_GRU, 128, j + 4, kb, lane));
            }
            v8f in_ = bbc(wb, BOFF_IN, j, lane);
            v8f hn  = bbc(wb, BOFF_HN, j, lane);
#pragma unroll
            for (int kb = 0; kb < 2; ++kb) {
                WMMA(in_, zA[kb], ldB(sW + OFF_IN, 64, j, kb, lane));
                WMMA(hn,  hA[kb], ldB(sW + OFF_HN, 64, j, kb, lane));
            }
            v8f rnn;
#pragma unroll
            for (int i = 0; i < 8; ++i) {
                float rv = sigf(r[i]);
                float uv = sigf(u[i]);
                float nv = tanhf_(in_[i] + rv * hn[i]);
                rnn[i] = (1.0f - uv) * nv + uv * hD[j][i];
            }
            hD[j] = rnn;                       // new hidden (D layout, f32)
            stAct(act, ZD, j, rnn, lane);      // bounce to LDS for A-layout reload
        }
        v16h rnnA[2];
        rnnA[0] = ldA(act, ZD, 0, lane);
        rnnA[1] = ldA(act, ZD, 1, lane);

        // ================= GatedTransition =================
        // hg = relu(rnn @ W_gzh^T + b)  [16,128]
#pragma unroll
        for (int j = 0; j < 8; ++j) {
            v8f c = bbc(wb, BOFF_GZH, j, lane);
            WMMA(c, rnnA[0], ldB(sW + OFF_GZH, 64, j, 0, lane));
            WMMA(c, rnnA[1], ldB(sW + OFF_GZH, 64, j, 1, lane));
#pragma unroll
            for (int i = 0; i < 8; ++i) c[i] = fmaxf(c[i], 0.0f);
            stAct(act, THD, j, c, lane);
        }
        v16h hgA[4];
#pragma unroll
        for (int kb = 0; kb < 4; ++kb) hgA[kb] = ldA(act, THD, kb, lane);
        v8f gate[4];
#pragma unroll
        for (int j = 0; j < 4; ++j) {
            v8f c = bbc(wb, BOFF_GHZ, j, lane);
#pragma unroll
            for (int kb = 0; kb < 4; ++kb) WMMA(c, hgA[kb], ldB(sW + OFF_GHZ, 128, j, kb, lane));
#pragma unroll
            for (int i = 0; i < 8; ++i) gate[j][i] = sigf(c[i]);
        }
        // hp = relu(rnn @ W_pzh^T + b)
#pragma unroll
        for (int j = 0; j < 8; ++j) {
            v8f c = bbc(wb, BOFF_PZH, j, lane);
            WMMA(c, rnnA[0], ldB(sW + OFF_PZH, 64, j, 0, lane));
            WMMA(c, rnnA[1], ldB(sW + OFF_PZH, 64, j, 1, lane));
#pragma unroll
            for (int i = 0; i < 8; ++i) c[i] = fmaxf(c[i], 0.0f);
            stAct(act, THD, j, c, lane);
        }
        v16h hpA[4];
#pragma unroll
        for (int kb = 0; kb < 4; ++kb) hpA[kb] = ldA(act, THD, kb, lane);
        // z = (1-gate)*lin(rnn,Wzloc) + gate*pm
#pragma unroll
        for (int j = 0; j < 4; ++j) {
            v8f pm = bbc(wb, BOFF_PHZ, j, lane);
#pragma unroll
            for (int kb = 0; kb < 4; ++kb) WMMA(pm, hpA[kb], ldB(sW + OFF_PHZ, 128, j, kb, lane));
            v8f l0 = bbc(wb, BOFF_ZLOC, j, lane);
            WMMA(l0, rnnA[0], ldB(sW + OFF_ZLOC, 64, j, 0, lane));
            WMMA(l0, rnnA[1], ldB(sW + OFF_ZLOC, 64, j, 1, lane));
            v8f zz;
#pragma unroll
            for (int i = 0; i < 8; ++i) zz[i] = (1.0f - gate[j][i]) * l0[i] + gate[j][i] * pm[i];
            stAct(act, ZD, j, zz, lane);
        }
        zA[0] = ldA(act, ZD, 0, lane);   // new z state
        zA[1] = ldA(act, ZD, 1, lane);
        hA[0] = rnnA[0];                 // new hidden in A layout
        hA[1] = rnnA[1];

        // ================= Observation heads =================
        float* obase = out + ((size_t)rowBase * TLEN + t) * (2 * XD);
        const int ccol = (lane & 15);
        const int mb   = (lane >> 4) * 8;
#pragma unroll
        for (int head = 0; head < 2; ++head) {
            const int offH = head ? OFF_OSH : OFF_OLH;
            const int offX = head ? OFF_OSX : OFF_OLX;
            const int bofH = head ? BOFF_OSH : BOFF_OLH;
            const int bofX = head ? BOFF_OSX : BOFF_OLX;
#pragma unroll
            for (int j = 0; j < 4; ++j) {
                v8f c = bbc(wb, bofH, j, lane);
                WMMA(c, zA[0], ldB(sW + offH, 64, j, 0, lane));
                WMMA(c, zA[1], ldB(sW + offH, 64, j, 1, lane));
#pragma unroll
                for (int i = 0; i < 8; ++i) c[i] = fmaxf(c[i], 0.0f);
                stAct(act, OHD, j, c, lane);
            }
            v16h hoA[2];
            hoA[0] = ldA(act, OHD, 0, lane);
            hoA[1] = ldA(act, OHD, 1, lane);
#pragma unroll
            for (int j = 0; j < 2; ++j) {
                v8f c = bbc(wb, bofX, j, lane);
                WMMA(c, hoA[0], ldB(sW + offX, 64, j, 0, lane));
                WMMA(c, hoA[1], ldB(sW + offX, 64, j, 1, lane));
#pragma unroll
                for (int i = 0; i < 8; ++i)
                    obase[(size_t)(mb + i) * TLEN * 64 + head * 32 + j * 16 + ccol] = fmaxf(c[i], 0.0f);
            }
        }
    }
}

extern "C" void kernel_launch(void* const* d_in, const int* in_sizes, int n_in,
                              void* d_out, int out_size, void* d_ws, size_t ws_size,
                              hipStream_t stream) {
    // setup_inputs order:
    // 0 mini_batch (unused), 1 z0, 2 h0, 3 W_ih, 4 W_hh, 5 b_ih, 6 b_hh,
    // 7 W_gzh, 8 b_gzh, 9 W_ghz, 10 b_ghz, 11 W_pzh, 12 b_pzh, 13 W_phz, 14 b_phz,
    // 15 W_zloc, 16 b_zloc, 17 W_zscale(unused), 18 b_zscale(unused),
    // 19 W_olh, 20 b_olh, 21 W_olx, 22 b_olx, 23 W_osh, 24 b_osh, 25 W_osx, 26 b_osx
    const float* z0   = (const float*)d_in[1];
    const float* h0   = (const float*)d_in[2];
    _Float16* wh = (_Float16*)d_ws;
    float*    wb = (float*)((char*)d_ws + (size_t)NWH * sizeof(_Float16));

    pack_weights<<<(NWH + NBF + 255) / 256, 256, 0, stream>>>(
        (const float*)d_in[3],  (const float*)d_in[4],  (const float*)d_in[5],  (const float*)d_in[6],
        (const float*)d_in[7],  (const float*)d_in[8],  (const float*)d_in[9],  (const float*)d_in[10],
        (const float*)d_in[11], (const float*)d_in[12], (const float*)d_in[13], (const float*)d_in[14],
        (const float*)d_in[15], (const float*)d_in[16],
        (const float*)d_in[19], (const float*)d_in[20], (const float*)d_in[21], (const float*)d_in[22],
        (const float*)d_in[23], (const float*)d_in[24], (const float*)d_in[25], (const float*)d_in[26],
        wh, wb);

    dmm_rollout<<<NBATCH / 32, 64, 0, stream>>>(z0, h0, wh, wb, (float*)d_out);
}
